// CrossAttn_30150670418234
// MI455X (gfx1250) — matrix operand order repaired and checked by reading, not compiled, not run on hardware
//
#include <hip/hip_runtime.h>

// ---------------------------------------------------------------------------
// Problem constants (match reference)
// ---------------------------------------------------------------------------
#define B_      4
#define SV_     2048
#define ST_     100
#define DM_     1024
#define H_      16
#define DK_     64
#define DFF_    4096
#define TD_     512
#define NL_     4
#define STP_    112      // keys padded to 7 * 16 for score tiles
#define KEYPAD_ 128      // keys padded to 4 * 32 for the PV contraction

typedef __attribute__((ext_vector_type(16))) __bf16 v16bf;
typedef __attribute__((ext_vector_type(8)))  __bf16 v8bf;
typedef __attribute__((ext_vector_type(8)))  float  v8f;

// ---------------------------------------------------------------------------
// Helpers
// ---------------------------------------------------------------------------
__device__ __forceinline__ v8bf zero8() {
    v8bf z;
#pragma unroll
    for (int i = 0; i < 8; ++i) z[i] = (__bf16)0.0f;
    return z;
}

// CDNA5 async global->LDS copy (16B per lane), tracked by ASYNCcnt.
// lds_addr: low 32 bits of a generic LDS pointer == LDS byte offset (AS3 value).
__device__ __forceinline__ void async_copy_b128(unsigned lds_addr, const void* gaddr) {
    asm volatile("global_load_async_to_lds_b128 %0, %1, off"
                 :: "v"(lds_addr), "v"(gaddr)
                 : "memory");
}
__device__ __forceinline__ void wait_async0() {
    asm volatile("s_wait_asynccnt 0x0" ::: "memory");
}
__device__ __forceinline__ unsigned lds_addr_of(const void* p) {
    return (unsigned)(uintptr_t)p;
}

// A-fragment (16x32 bf16, M x K): lane m=l&15 is the row; kh = (l>>4)*8.
// vector elems 0..7  = K[kh .. kh+7], elems 8..15 = K[kh+16 .. kh+23]
__device__ __forceinline__ v16bf frag_a(const __bf16* base, int ld) {
    const int l = threadIdx.x & 31;
    const __bf16* p = base + (size_t)(l & 15) * ld + ((l >> 4) << 3);
    v8bf lo = *(const v8bf*)p;
    v8bf hi = *(const v8bf*)(p + 16);
    v16bf r;
#pragma unroll
    for (int i = 0; i < 8; ++i) { r[i] = lo[i]; r[i + 8] = hi[i]; }
    return r;
}

// B-fragment (32x16 bf16, K x N) read from a TRANSPOSED [N][K] tile:
// lane n=l&15 is the column; kh = (l>>4)*8; elems 0..7 = K[kh..kh+7], elems 8..15 = K[kh+16..kh+23]
__device__ __forceinline__ v16bf frag_bT(const __bf16* base, int ld) {
    const int l = threadIdx.x & 31;
    const __bf16* p = base + (size_t)(l & 15) * ld + ((l >> 4) << 3);
    v8bf lo = *(const v8bf*)p;
    v8bf hi = *(const v8bf*)(p + 16);
    v16bf r;
#pragma unroll
    for (int i = 0; i < 8; ++i) { r[i] = lo[i]; r[i + 8] = hi[i]; }
    return r;
}

__device__ __forceinline__ v8f wmma_bf16(v16bf a, v16bf b, v8f c) {
    return __builtin_amdgcn_wmma_f32_16x16x32_bf16(
        /*neg_a=*/false, a, /*neg_b=*/false, b,
        /*c_mod=*/(short)0, c, /*reuse_a=*/false, /*reuse_b=*/false);
}

// ---------------------------------------------------------------------------
// Positional-encoding add kernels
// ---------------------------------------------------------------------------
__global__ __launch_bounds__(256) void pe_video_kernel(
    const float* __restrict__ X, float* __restrict__ Yf, __bf16* __restrict__ Yb) {
    size_t i = (size_t)blockIdx.x * 256 + threadIdx.x;
    int d = (int)(i & (DM_ - 1));
    int s = (int)((i >> 10) & (SV_ - 1));
    int j2 = (d >> 1) << 1;
    float f = __expf(-(float)j2 * (9.210340371976184f / (float)DM_));
    float ang = (float)s * f;
    float pe = (d & 1) ? __cosf(ang) : __sinf(ang);
    float v = X[i] + pe;
    Yf[i] = v;
    Yb[i] = (__bf16)v;
}

__global__ __launch_bounds__(256) void pe_text_kernel(
    const float* __restrict__ X, __bf16* __restrict__ Yb, int total) {
    int i = blockIdx.x * 256 + threadIdx.x;
    if (i >= total) return;
    int d = i & (TD_ - 1);
    int s = (i >> 9) % ST_;
    int j2 = (d >> 1) << 1;
    float f = __expf(-(float)j2 * (9.210340371976184f / (float)TD_));
    float ang = (float)s * f;
    float pe = (d & 1) ? __cosf(ang) : __sinf(ang);
    Yb[i] = (__bf16)(X[i] + pe);
}

// ---------------------------------------------------------------------------
// Weight convert + transpose: fp32 [K,N] -> bf16 [N,K]
// ---------------------------------------------------------------------------
__global__ __launch_bounds__(256) void wtrans_kernel(
    const float* __restrict__ W, __bf16* __restrict__ Wt, int K, int N) {
    __shared__ __bf16 t[32][33];
    const int nb = blockIdx.x * 32, kb = blockIdx.y * 32;
    const int tx = threadIdx.x & 31, ty = threadIdx.x >> 5;
#pragma unroll
    for (int r = 0; r < 4; ++r) {
        int k = ty + r * 8;
        t[k][tx] = (__bf16)W[(size_t)(kb + k) * N + nb + tx];
    }
    __syncthreads();
#pragma unroll
    for (int r = 0; r < 4; ++r) {
        int n = ty + r * 8;
        Wt[(size_t)(nb + n) * K + kb + tx] = t[tx][n];
    }
}

// ---------------------------------------------------------------------------
// bf16 WMMA GEMM: C[M,N] = A[M,K] * Bt[N,K]^T + bias (+resid) (relu) -> f32/bf16
// Block tile 128x128, 8 waves, wave tile 64x32, BK=32.
// Tile staging uses CDNA5 async global->LDS copies (ASYNCcnt), double-buffered.
// ---------------------------------------------------------------------------
template <bool RELU, bool RESID, bool BF16OUT>
__global__ __launch_bounds__(256) void gemm_bf16_kernel(
    const __bf16* __restrict__ A, const __bf16* __restrict__ Bt,
    const float* __restrict__ bias, const float* __restrict__ resid,
    float* __restrict__ Cf, __bf16* __restrict__ Cb,
    int M, int N, int K) {
    constexpr int BM = 128, BN = 128, BK = 32, LDT = BK + 8;
    __shared__ __bf16 sA[2][BM][LDT];
    __shared__ __bf16 sB[2][BN][LDT];

    const int tid = threadIdx.x;
    const int ldRow = tid >> 2;            // 0..63
    const int ldCol = (tid & 3) << 3;      // 0,8,16,24 halves
    const int rowBase = blockIdx.y * BM;
    const int colBase = blockIdx.x * BN;
    const int wave = tid >> 5;
    const int wm = (wave >> 2) * 64;
    const int wn = (wave & 3) * 32;
    const int lane = tid & 31;
    const int lhi = lane >> 4, lln = lane & 15;

    v8f acc[4][2];
#pragma unroll
    for (int mi = 0; mi < 4; ++mi)
#pragma unroll
        for (int ni = 0; ni < 2; ++ni)
#pragma unroll
            for (int r = 0; r < 8; ++r) acc[mi][ni][r] = 0.0f;

    // Clamp OOB A rows to M-1: a tile row only feeds its own (discarded) C row.
    const int gr0 = min(rowBase + ldRow, M - 1);
    const int gr1 = min(rowBase + ldRow + 64, M - 1);
    const int gn0 = colBase + ldRow;
    const __bf16* aRow0 = A + (size_t)gr0 * K + ldCol;
    const __bf16* aRow1 = A + (size_t)gr1 * K + ldCol;
    const __bf16* bRow0 = Bt + (size_t)gn0 * K + ldCol;
    const __bf16* bRow1 = Bt + (size_t)(gn0 + 64) * K + ldCol;

    const unsigned la0 = lds_addr_of(&sA[0][ldRow][ldCol]);
    const unsigned la1 = lds_addr_of(&sA[0][ldRow + 64][ldCol]);
    const unsigned lb0 = lds_addr_of(&sB[0][ldRow][ldCol]);
    const unsigned lb1 = lds_addr_of(&sB[0][ldRow + 64][ldCol]);
    const unsigned bufStride = (unsigned)(sizeof(__bf16) * BM * LDT);

    auto stage = [&](int buf, int k0) {
        const unsigned o = buf ? bufStride : 0u;
        async_copy_b128(la0 + o, aRow0 + k0);
        async_copy_b128(la1 + o, aRow1 + k0);
        async_copy_b128(lb0 + o, bRow0 + k0);
        async_copy_b128(lb1 + o, bRow1 + k0);
    };

    const int kTiles = K / BK;
    stage(0, 0);
    wait_async0();
    __syncthreads();

    for (int kt = 0; kt < kTiles; ++kt) {
        const int cur = kt & 1;
        const bool more = (kt + 1) < kTiles;
        if (more) stage(cur ^ 1, (kt + 1) * BK);   // overlap with compute below

        v16bf afr[4], bfr[2];
#pragma unroll
        for (int mi = 0; mi < 4; ++mi) afr[mi] = frag_a(&sA[cur][wm + mi * 16][0], LDT);
#pragma unroll
        for (int ni = 0; ni < 2; ++ni) bfr[ni] = frag_bT(&sB[cur][wn + ni * 16][0], LDT);
#pragma unroll
        for (int mi = 0; mi < 4; ++mi)
#pragma unroll
            for (int ni = 0; ni < 2; ++ni)
                acc[mi][ni] = wmma_bf16(afr[mi], bfr[ni], acc[mi][ni]);

        if (more) {
            wait_async0();
            __syncthreads();
        }
    }

    // epilogue: C layout reg r -> row r + 8*(lane>=16), col = lane&15
#pragma unroll
    for (int mi = 0; mi < 4; ++mi) {
#pragma unroll
        for (int ni = 0; ni < 2; ++ni) {
            const int col = colBase + wn + ni * 16 + lln;
            const float bv = bias[col];
#pragma unroll
            for (int r = 0; r < 8; ++r) {
                const int row = rowBase + wm + mi * 16 + r + 8 * lhi;
                if (row < M) {
                    float c = acc[mi][ni][r] + bv;
                    if (RESID) c += resid[(size_t)row * N + col];
                    if (RELU) c = fmaxf(c, 0.0f);
                    if (BF16OUT) Cb[(size_t)row * N + col] = (__bf16)c;
                    else         Cf[(size_t)row * N + col] = c;
                }
            }
        }
    }
}

// ---------------------------------------------------------------------------
// Cross-attention: grid (SV/64, H, B), 128 threads (4 waves, 16 queries/wave).
// S = Q K^T / 8, mask, softmax over ST keys, O = P V.  All matmuls via WMMA.
// ---------------------------------------------------------------------------
__global__ __launch_bounds__(128) void attn_kernel(
    const __bf16* __restrict__ Q, const __bf16* __restrict__ Kc,
    const __bf16* __restrict__ Vc, const int* __restrict__ mask,
    __bf16* __restrict__ O) {
    constexpr int LDK = 72;            // sK row stride (halves)
    constexpr int LDV = KEYPAD_ + 8;   // 136 halves
    __shared__ __bf16 sK[STP_][LDK];       // [key][dk]
    __shared__ __bf16 sV[DK_][LDV];        // [dk][key] (transposed)
    __shared__ __bf16 sP[4][16][LDV];      // per-wave probabilities

    const int tid = threadIdx.x;
    const int b = blockIdx.z, h = blockIdx.y;
    const int lane = tid & 31, w = tid >> 5;
    const int lhi = lane >> 4, lln = lane & 15;

    // stage K tile [key][dk]: async copies for valid keys, zero-fill the pad
    for (int c = tid; c < STP_ * 8; c += 128) {
        int key = c >> 3, d8 = (c & 7) << 3;
        if (key < ST_) {
            async_copy_b128(lds_addr_of(&sK[key][d8]),
                            Kc + (size_t)(b * ST_ + key) * DM_ + h * DK_ + d8);
        } else {
            *(v8bf*)&sK[key][d8] = zero8();
        }
    }
    // stage V transposed [dk][key]
    for (int c = tid; c < STP_ * 8; c += 128) {
        int key = c >> 3, d8 = (c & 7) << 3;
        v8bf val = zero8();
        if (key < ST_) val = *(const v8bf*)(Vc + (size_t)(b * ST_ + key) * DM_ + h * DK_ + d8);
#pragma unroll
        for (int i = 0; i < 8; ++i) sV[d8 + i][key] = val[i];
    }
    for (int c = tid; c < DK_ * (KEYPAD_ - STP_); c += 128) {
        int d = c >> 4, key = STP_ + (c & 15);
        sV[d][key] = (__bf16)0.0f;
    }
    wait_async0();
    __syncthreads();

    const int q0 = blockIdx.x * 64 + w * 16;
    const __bf16* qbase = Q + (size_t)(b * SV_ + q0) * DM_ + h * DK_;

    // scores: 7 key tiles of 16, dk contraction split into 2 k-steps of 32
    v8f sc[7];
#pragma unroll
    for (int t = 0; t < 7; ++t)
#pragma unroll
        for (int r = 0; r < 8; ++r) sc[t][r] = 0.0f;
#pragma unroll
    for (int ks = 0; ks < 2; ++ks) {
        v16bf qa = frag_a(qbase + ks * 32, DM_);
#pragma unroll
        for (int t = 0; t < 7; ++t) {
            v16bf kb = frag_bT(&sK[t * 16][ks * 32], LDK);
            sc[t] = wmma_bf16(qa, kb, sc[t]);
        }
    }

    // mask + scale + softmax (rows live across the 16-lane half)
    const int* mb = mask + b * ST_;
    float mx[8], sm[8];
#pragma unroll
    for (int r = 0; r < 8; ++r) { mx[r] = -3.0e38f; sm[r] = 0.0f; }
#pragma unroll
    for (int t = 0; t < 7; ++t) {
        int key = t * 16 + lln;
        bool valid = false;
        if (key < ST_) valid = (mb[key] != 0);
#pragma unroll
        for (int r = 0; r < 8; ++r) {
            float s = sc[t][r] * 0.125f;
            s = valid ? s : -1.0e9f;
            sc[t][r] = s;
            mx[r] = fmaxf(mx[r], s);
        }
    }
#pragma unroll
    for (int off = 1; off < 16; off <<= 1)
#pragma unroll
        for (int r = 0; r < 8; ++r) mx[r] = fmaxf(mx[r], __shfl_xor(mx[r], off, 32));
#pragma unroll
    for (int t = 0; t < 7; ++t)
#pragma unroll
        for (int r = 0; r < 8; ++r) {
            float p = __expf(sc[t][r] - mx[r]);
            sc[t][r] = p;
            sm[r] += p;
        }
#pragma unroll
    for (int off = 1; off < 16; off <<= 1)
#pragma unroll
        for (int r = 0; r < 8; ++r) sm[r] += __shfl_xor(sm[r], off, 32);

    // stage P (C-layout -> LDS -> A-fragment layout), zero padded keys
#pragma unroll
    for (int t = 0; t < 7; ++t)
#pragma unroll
        for (int r = 0; r < 8; ++r)
            sP[w][r + 8 * lhi][t * 16 + lln] = (__bf16)sc[t][r];
#pragma unroll
    for (int r = 0; r < 8; ++r)
        sP[w][r + 8 * lhi][STP_ + lln] = (__bf16)0.0f;
    __syncthreads();

    // O = P V : M=16 queries, N=64 dk (4 tiles), K=128 keys (4 steps)
    v8f oa[4];
#pragma unroll
    for (int n = 0; n < 4; ++n)
#pragma unroll
        for (int r = 0; r < 8; ++r) oa[n][r] = 0.0f;
#pragma unroll
    for (int kt = 0; kt < 4; ++kt) {
        v16bf pa = frag_a(&sP[w][0][kt * 32], LDV);
#pragma unroll
        for (int n = 0; n < 4; ++n) {
            v16bf vb = frag_bT(&sV[n * 16][kt * 32], LDV);
            oa[n] = wmma_bf16(pa, vb, oa[n]);
        }
    }

    float inv[8];
#pragma unroll
    for (int r = 0; r < 8; ++r) inv[r] = 1.0f / sm[r];
#pragma unroll
    for (int n = 0; n < 4; ++n) {
        int dc = n * 16 + lln;
#pragma unroll
        for (int r = 0; r < 8; ++r) {
            int qr = q0 + r + 8 * lhi;
            O[(size_t)(b * SV_ + qr) * DM_ + h * DK_ + dc] = (__bf16)(oa[n][r] * inv[r]);
        }
    }
}

// ---------------------------------------------------------------------------
// LayerNorm over D=1024, one block per row; writes f32 + bf16 copies.
// ---------------------------------------------------------------------------
__global__ __launch_bounds__(256) void layernorm_kernel(
    const float* __restrict__ X, const float* __restrict__ g,
    const float* __restrict__ be, float* __restrict__ Yf, __bf16* __restrict__ Yb) {
    const size_t row = blockIdx.x;
    const float* x = X + row * DM_;
    float v[4], s = 0.0f, s2 = 0.0f;
#pragma unroll
    for (int i = 0; i < 4; ++i) {
        v[i] = x[threadIdx.x + i * 256];
        s += v[i];
        s2 += v[i] * v[i];
    }
#pragma unroll
    for (int off = 16; off > 0; off >>= 1) {
        s += __shfl_xor(s, off, 32);
        s2 += __shfl_xor(s2, off, 32);
    }
    __shared__ float rs[8], rs2[8];
    const int w = threadIdx.x >> 5, l = threadIdx.x & 31;
    if (l == 0) { rs[w] = s; rs2[w] = s2; }
    __syncthreads();
    s = 0.0f; s2 = 0.0f;
#pragma unroll
    for (int i = 0; i < 8; ++i) { s += rs[i]; s2 += rs2[i]; }
    const float mu = s * (1.0f / DM_);
    const float var = s2 * (1.0f / DM_) - mu * mu;
    const float rstd = rsqrtf(var + 1e-5f);
#pragma unroll
    for (int i = 0; i < 4; ++i) {
        int c = threadIdx.x + i * 256;
        float y = (v[i] - mu) * rstd * g[c] + be[c];
        Yf[row * DM_ + c] = y;
        Yb[row * DM_ + c] = (__bf16)y;
    }
}

// ---------------------------------------------------------------------------
// Launcher
// ---------------------------------------------------------------------------
extern "C" void kernel_launch(void* const* d_in, const int* in_sizes, int n_in,
                              void* d_out, int out_size, void* d_ws, size_t ws_size,
                              hipStream_t stream) {
    (void)in_sizes; (void)n_in; (void)out_size; (void)ws_size;

    const float* video = (const float*)d_in[0];
    const float* text  = (const float*)d_in[1];
    const int*   tmask = (const int*)d_in[2];
    const float* Wq_w = (const float*)d_in[3];
    const float* Wq_b = (const float*)d_in[4];
    const float* Wk_w = (const float*)d_in[5];
    const float* Wk_b = (const float*)d_in[6];
    const float* Wv_w = (const float*)d_in[7];
    const float* Wv_b = (const float*)d_in[8];
    const float* Wo_w = (const float*)d_in[9];
    const float* Wo_b = (const float*)d_in[10];
    const float* f1_w = (const float*)d_in[11];
    const float* f1_b = (const float*)d_in[12];
    const float* f2_w = (const float*)d_in[13];
    const float* f2_b = (const float*)d_in[14];
    const float* n2g  = (const float*)d_in[15];
    const float* n2b  = (const float*)d_in[16];
    const float* n3g  = (const float*)d_in[17];
    const float* n3b  = (const float*)d_in[18];

    char* ws = (char*)d_ws;
    size_t off = 0;
    auto alloc = [&](size_t bytes) -> char* {
        char* p = ws + off;
        off += (bytes + 255) & ~(size_t)255;
        return p;
    };
    float*  Vcur = (float*)alloc((size_t)B_ * SV_ * DM_ * 4);   // fp32 residual stream
    float*  T32  = (float*)alloc((size_t)B_ * SV_ * DM_ * 4);   // pre-LN temp
    __bf16* Vbf  = (__bf16*)alloc((size_t)B_ * SV_ * DM_ * 2);  // bf16 activation
    __bf16* Qbf  = (__bf16*)alloc((size_t)B_ * SV_ * DM_ * 2);
    __bf16* Abf  = (__bf16*)alloc((size_t)B_ * SV_ * DM_ * 2);
    __bf16* Hbf  = (__bf16*)alloc((size_t)B_ * SV_ * DFF_ * 2);
    __bf16* Tbf  = (__bf16*)alloc((size_t)B_ * ST_ * TD_ * 2);
    __bf16* Kbf  = (__bf16*)alloc((size_t)B_ * ST_ * DM_ * 2);
    __bf16* Vvbf = (__bf16*)alloc((size_t)B_ * ST_ * DM_ * 2);
    __bf16* WqT  = (__bf16*)alloc((size_t)DM_ * DM_ * 2);
    __bf16* WkT  = (__bf16*)alloc((size_t)TD_ * DM_ * 2);
    __bf16* WvT  = (__bf16*)alloc((size_t)TD_ * DM_ * 2);
    __bf16* WoT  = (__bf16*)alloc((size_t)DM_ * DM_ * 2);
    __bf16* F1T  = (__bf16*)alloc((size_t)DM_ * DFF_ * 2);
    __bf16* F2T  = (__bf16*)alloc((size_t)DFF_ * DM_ * 2);

    const int MV = B_ * SV_;   // 8192
    const int MT = B_ * ST_;   // 400

    // positional encodings
    pe_video_kernel<<<dim3((unsigned)((size_t)MV * DM_ / 256)), 256, 0, stream>>>(video, Vcur, Vbf);
    {
        int tot = B_ * ST_ * TD_;
        pe_text_kernel<<<dim3((tot + 255) / 256), 256, 0, stream>>>(text, Tbf, tot);
    }

    for (int l = 0; l < NL_; ++l) {
        // transpose+convert this layer's weights to bf16 [N,K]
        wtrans_kernel<<<dim3(DM_ / 32, DM_ / 32), 256, 0, stream>>>(Wq_w + (size_t)l * DM_ * DM_, WqT, DM_, DM_);
        wtrans_kernel<<<dim3(DM_ / 32, TD_ / 32), 256, 0, stream>>>(Wk_w + (size_t)l * TD_ * DM_, WkT, TD_, DM_);
        wtrans_kernel<<<dim3(DM_ / 32, TD_ / 32), 256, 0, stream>>>(Wv_w + (size_t)l * TD_ * DM_, WvT, TD_, DM_);
        wtrans_kernel<<<dim3(DM_ / 32, DM_ / 32), 256, 0, stream>>>(Wo_w + (size_t)l * DM_ * DM_, WoT, DM_, DM_);
        wtrans_kernel<<<dim3(DFF_ / 32, DM_ / 32), 256, 0, stream>>>(f1_w + (size_t)l * DM_ * DFF_, F1T, DM_, DFF_);
        wtrans_kernel<<<dim3(DM_ / 32, DFF_ / 32), 256, 0, stream>>>(f2_w + (size_t)l * DFF_ * DM_, F2T, DFF_, DM_);

        // Q = Vbf @ Wq + bq  -> bf16
        gemm_bf16_kernel<false, false, true><<<dim3(DM_ / 128, MV / 128), 256, 0, stream>>>(
            Vbf, WqT, Wq_b + (size_t)l * DM_, nullptr, nullptr, Qbf, MV, DM_, DM_);
        // K,V = Tbf @ Wk/Wv + b -> bf16
        gemm_bf16_kernel<false, false, true><<<dim3(DM_ / 128, (MT + 127) / 128), 256, 0, stream>>>(
            Tbf, WkT, Wk_b + (size_t)l * DM_, nullptr, nullptr, Kbf, MT, DM_, TD_);
        gemm_bf16_kernel<false, false, true><<<dim3(DM_ / 128, (MT + 127) / 128), 256, 0, stream>>>(
            Tbf, WvT, Wv_b + (size_t)l * DM_, nullptr, nullptr, Vvbf, MT, DM_, TD_);

        // attention -> bf16
        attn_kernel<<<dim3(SV_ / 64, H_, B_), 128, 0, stream>>>(Qbf, Kbf, Vvbf, tmask, Abf);

        // O-proj + residual -> fp32 temp ; LN2 -> Vcur(f32) + Vbf(bf16)
        gemm_bf16_kernel<false, true, false><<<dim3(DM_ / 128, MV / 128), 256, 0, stream>>>(
            Abf, WoT, Wo_b + (size_t)l * DM_, Vcur, T32, nullptr, MV, DM_, DM_);
        layernorm_kernel<<<dim3(MV), 256, 0, stream>>>(
            T32, n2g + (size_t)l * DM_, n2b + (size_t)l * DM_, Vcur, Vbf);

        // FFN: fc1 (ReLU, bf16 out), fc2 (+residual, f32 out), LN3
        gemm_bf16_kernel<true, false, true><<<dim3(DFF_ / 128, MV / 128), 256, 0, stream>>>(
            Vbf, F1T, f1_b + (size_t)l * DFF_, nullptr, nullptr, Hbf, MV, DFF_, DM_);
        gemm_bf16_kernel<false, true, false><<<dim3(DM_ / 128, MV / 128), 256, 0, stream>>>(
            Hbf, F2T, f2_b + (size_t)l * DM_, Vcur, T32, nullptr, MV, DM_, DFF_);
        float* outp = (l == NL_ - 1) ? (float*)d_out : Vcur;
        layernorm_kernel<<<dim3(MV), 256, 0, stream>>>(
            T32, n3g + (size_t)l * DM_, n3b + (size_t)l * DM_, outp, Vbf);
    }
}
